// QModel_33449205301621
// MI455X (gfx1250) — matrix operand order repaired and checked by reading, not compiled, not run on hardware
//
#include <hip/hip_runtime.h>
#include <hip/hip_bf16.h>

typedef __bf16 bf16;
typedef __attribute__((ext_vector_type(8)))  bf16  v8bf;
typedef __attribute__((ext_vector_type(16))) bf16  v16bf;
typedef __attribute__((ext_vector_type(8)))  float v8f;

union V16 { v16bf v; v8bf h[2]; };

#define E    512
#define BM   64
#define ACOL 4096

#if __has_builtin(__builtin_amdgcn_global_load_async_to_lds_b128)
#define HAS_ASYNC 1
#else
#define HAS_ASYNC 0
#endif

__device__ __forceinline__ bf16 f2bf(float f) {
    union { float f; unsigned u; } c; c.f = f;
    unsigned r = c.u + 0x7fffu + ((c.u >> 16) & 1u);   // round-to-nearest-even
    union { unsigned short s; bf16 b; } o; o.s = (unsigned short)(r >> 16);
    return o.b;
}

// 16-byte global -> LDS copy; async path on gfx1250 toolchains that expose it.
__device__ __forceinline__ void cp16_g2l(const void* g, void* l) {
#if HAS_ASYNC
    typedef int v4i_async __attribute__((vector_size(16)));
    __builtin_amdgcn_global_load_async_to_lds_b128(
        (__attribute__((address_space(1))) v4i_async*)g,
        (__attribute__((address_space(3))) v4i_async*)l, 0, 0);
#else
    *(uint4*)l = *(const uint4*)g;
#endif
}

__device__ __forceinline__ void cp_wait() {
#if HAS_ASYNC
#if __has_builtin(__builtin_amdgcn_s_wait_asynccnt)
    __builtin_amdgcn_s_wait_asynccnt(0);
#else
    asm volatile("s_wait_asynccnt 0" ::: "memory");
#endif
#endif
}

// Stage BM x E bf16 rows into LDS: async copy from pre-converted Xbf if
// available, otherwise load f32 X and convert in-register.
__device__ __forceinline__ void stage_X(const float* __restrict__ X,
                                        const bf16* __restrict__ Xbf,
                                        bf16 (*Xs)[E], int row0, int N, int tid) {
    if (Xbf) {
        const size_t limit = (size_t)N * (E * 2);
        char* lbase = (char*)&Xs[0][0];
        for (int c = tid; c < (BM * E * 2) / 16; c += 256) {
            size_t goff = (size_t)row0 * (E * 2) + (size_t)c * 16;
            if (goff < limit) cp16_g2l((const char*)Xbf + goff, lbase + c * 16);
        }
        cp_wait();
    } else {
        for (int idx = tid; idx < BM * E; idx += 256) {
            int m = idx >> 9, k = idx & (E - 1);
            float x = (row0 + m < N) ? X[(size_t)(row0 + m) * E + k] : 0.f;
            Xs[m][k] = f2bf(x);
        }
    }
}

// ---------------------------------------------------------------------------
// f32 (K x Ncols row-major) -> bf16 transposed (Ncols x K): B-fragments become
// one contiguous 32B run per lane.
// ---------------------------------------------------------------------------
__global__ void k_convT(const float* __restrict__ W, bf16* __restrict__ Wt,
                        int K, int Ncols) {
    int idx = blockIdx.x * blockDim.x + threadIdx.x;
    if (idx >= K * Ncols) return;
    int k = idx / Ncols, n = idx - k * Ncols;
    Wt[(size_t)n * K + k] = f2bf(W[idx]);
}

// f32 -> bf16 flat (for X; keeps the whole bf16 working set L2-resident)
__global__ void k_convX(const float* __restrict__ X, bf16* __restrict__ Xbf,
                        size_t total) {
    size_t idx = (size_t)blockIdx.x * blockDim.x + threadIdx.x;
    if (idx < total) Xbf[idx] = f2bf(X[idx]);
}

// ---------------------------------------------------------------------------
// device_q = relu(X @ W1d + b1d) @ W2d + b2d        (N x 2)
// 64 rows/block, 8 waves = 4 row-blocks x 2 col-strips (16 wmma tiles each).
// The 512->2 projection is fused: shuffle reductions + LDS atomics into a
// scratch region aliased over the consumed X tile (LDS stays at 64KB).
// ---------------------------------------------------------------------------
__global__ __launch_bounds__(256) void k_device(
    const float* __restrict__ X, const bf16* __restrict__ Xbf,
    const bf16* __restrict__ Wt1, const float* __restrict__ b1,
    const float* __restrict__ W2, const float* __restrict__ b2,
    float* __restrict__ dq, int N) {
    __shared__ bf16 Xs[BM][E];                 // 64 KB
    float* dqs = (float*)&Xs[0][0];            // aliased after X is consumed
    const int tid = threadIdx.x, lane = tid & 31, w = tid >> 5;
    const int row0 = blockIdx.x * BM;

    stage_X(X, Xbf, Xs, row0, N, tid);
    __syncthreads();

    const int mb  = (w >> 1) * 16;             // row-block
    const int nb0 = (w & 1) * 256;             // col-strip
    const int h16 = lane >> 4, l15 = lane & 15;

    v8f acc[16];
    for (int t = 0; t < 16; ++t) for (int j = 0; j < 8; ++j) acc[t][j] = 0.f;

    for (int kb = 0; kb < E; kb += 32) {
        V16 a;
        a.h[0] = *(const v8bf*)&Xs[mb + l15][kb + 16 * h16];
        a.h[1] = *(const v8bf*)&Xs[mb + l15][kb + 16 * h16 + 8];
        for (int t = 0; t < 16; ++t) {
            int n = nb0 + t * 16 + l15;
            V16 bm;
            bm.h[0] = *(const v8bf*)&Wt1[(size_t)n * E + kb + 16 * h16];
            bm.h[1] = *(const v8bf*)&Wt1[(size_t)n * E + kb + 16 * h16 + 8];
            acc[t] = __builtin_amdgcn_wmma_f32_16x16x32_bf16(
                false, a.v, false, bm.v, (short)0, acc[t], false, false);
        }
    }

    float p0[8], p1[8];
    for (int i = 0; i < 8; ++i) { p0[i] = 0.f; p1[i] = 0.f; }
    for (int t = 0; t < 16; ++t) {
        int n = nb0 + t * 16 + l15;
        float w20 = W2[n * 2 + 0], w21 = W2[n * 2 + 1], bb = b1[n];
        for (int i = 0; i < 8; ++i) {
            float hv = acc[t][i] + bb; hv = hv > 0.f ? hv : 0.f;
            p0[i] += hv * w20; p1[i] += hv * w21;
        }
    }
    __syncthreads();                            // all waves done with Xs
    if (tid < 2 * BM) dqs[tid] = 0.f;
    __syncthreads();
    for (int i = 0; i < 8; ++i) {
        for (int mask = 1; mask <= 8; mask <<= 1) {
            p0[i] += __shfl_xor(p0[i], mask, 32);
            p1[i] += __shfl_xor(p1[i], mask, 32);
        }
        if (l15 == 0) {
            int m = mb + i + 8 * h16;
            atomicAdd(&dqs[m * 2 + 0], p0[i]);
            atomicAdd(&dqs[m * 2 + 1], p1[i]);
        }
    }
    __syncthreads();
    if (tid < 2 * BM) {
        int m = tid >> 1, j = tid & 1;
        if (row0 + m < N) dq[(size_t)(row0 + m) * 2 + j] = dqs[tid] + b2[j];
    }
}

// ---------------------------------------------------------------------------
// seg[batch] = sum over segment of device_q[:,1]
// ---------------------------------------------------------------------------
__global__ void k_seg(const float* __restrict__ dq, const int* __restrict__ si,
                      float* __restrict__ seg) {
    int batch = blockIdx.x, lane = threadIdx.x;
    int s0 = si[batch], s1 = si[batch + 1];
    float v = 0.f;
    for (int i = s0 + lane; i < s1; i += 32) v += dq[(size_t)i * 2 + 1];
    for (int mask = 16; mask >= 1; mask >>= 1) v += __shfl_xor(v, mask, 32);
    if (lane == 0) seg[batch] = v;
}

// ---------------------------------------------------------------------------
// Fused action path per 64-row block:
//   Ha = relu(X @ W1a + b1a)            (WMMA; result overwrites the X tile)
//   A  = Ha @ W2a + b2a + rowbias       (WMMA; 16 x 2048 strip per wave)
// scattered straight into out[batch*max_d + dev_idx, :].
// Row bias / output row kept in registers (8 per lane) -> LDS stays 64KB.
// ---------------------------------------------------------------------------
__global__ __launch_bounds__(256) void k_action(
    const float* __restrict__ X, const bf16* __restrict__ Xbf,
    const bf16* __restrict__ Wt1, const float* __restrict__ b1,
    const bf16* __restrict__ Wt2, const float* __restrict__ b2,
    const float* __restrict__ dq, const float* __restrict__ seg,
    const int* __restrict__ bidx, const int* __restrict__ sidx,
    float* __restrict__ out, int N, int max_d) {
    __shared__ bf16 Xs[BM][E];                 // 64 KB; becomes Ha after GEMM1
    const int tid = threadIdx.x, lane = tid & 31, w = tid >> 5;
    const int row0 = blockIdx.x * BM;

    stage_X(X, Xbf, Xs, row0, N, tid);
    __syncthreads();

    const int mb  = (w >> 1) * 16;             // row-block (0..3)
    const int h16 = lane >> 4, l15 = lane & 15;

    { // GEMM1: relu(X @ W1a + b1a) -> Xs (bf16)
        const int nb0 = (w & 1) * 256;
        v8f acc[16];
        for (int t = 0; t < 16; ++t) for (int j = 0; j < 8; ++j) acc[t][j] = 0.f;
        for (int kb = 0; kb < E; kb += 32) {
            V16 a;
            a.h[0] = *(const v8bf*)&Xs[mb + l15][kb + 16 * h16];
            a.h[1] = *(const v8bf*)&Xs[mb + l15][kb + 16 * h16 + 8];
            for (int t = 0; t < 16; ++t) {
                int n = nb0 + t * 16 + l15;
                V16 bm;
                bm.h[0] = *(const v8bf*)&Wt1[(size_t)n * E + kb + 16 * h16];
                bm.h[1] = *(const v8bf*)&Wt1[(size_t)n * E + kb + 16 * h16 + 8];
                acc[t] = __builtin_amdgcn_wmma_f32_16x16x32_bf16(
                    false, a.v, false, bm.v, (short)0, acc[t], false, false);
            }
        }
        __syncthreads();                       // all waves done reading X
        for (int t = 0; t < 16; ++t) {
            int n = nb0 + t * 16 + l15;
            float bb = b1[n];
            for (int i = 0; i < 8; ++i) {
                float hv = acc[t][i] + bb; hv = hv > 0.f ? hv : 0.f;
                Xs[mb + i + 8 * h16][n] = f2bf(hv);
            }
        }
        __syncthreads();
    }

    // Per-lane row bias + output row for this lane's 8 output rows.
    float rbv[8];
    int   orv[8];
    for (int i = 0; i < 8; ++i) {
        int gr = row0 + mb + i + 8 * h16;
        if (gr < N) {
            int bi = bidx[gr];
            int s0 = sidx[bi], s1 = sidx[bi + 1];
            float d0 = dq[(size_t)gr * 2 + 0], d1 = dq[(size_t)gr * 2 + 1];
            rbv[i] = d0 + (seg[bi] - d1) / ((float)(s1 - s0) - 1.0f + 1e-8f);
            orv[i] = bi * max_d + (gr - s0);
        } else { rbv[i] = 0.f; orv[i] = -1; }
    }

    { // GEMM2: Ha @ W2a, 16 x 2048 strip per wave, 64-col chunks
        const int cs0 = (w & 1) * 2048;
        for (int cc = 0; cc < 2048; cc += 64) {
            int nbase = cs0 + cc;
            v8f acc[4];
            for (int t = 0; t < 4; ++t) for (int j = 0; j < 8; ++j) acc[t][j] = 0.f;
            for (int kb = 0; kb < E; kb += 32) {
                V16 a;
                a.h[0] = *(const v8bf*)&Xs[mb + l15][kb + 16 * h16];
                a.h[1] = *(const v8bf*)&Xs[mb + l15][kb + 16 * h16 + 8];
                for (int t = 0; t < 4; ++t) {
                    int n = nbase + t * 16 + l15;
                    V16 bm;
                    bm.h[0] = *(const v8bf*)&Wt2[(size_t)n * E + kb + 16 * h16];
                    bm.h[1] = *(const v8bf*)&Wt2[(size_t)n * E + kb + 16 * h16 + 8];
                    acc[t] = __builtin_amdgcn_wmma_f32_16x16x32_bf16(
                        false, a.v, false, bm.v, (short)0, acc[t], false, false);
                }
            }
            for (int t = 0; t < 4; ++t) {
                int n = nbase + t * 16 + l15;
                float bcol = b2[n];
                for (int i = 0; i < 8; ++i) {
                    if (orv[i] >= 0)
                        out[(size_t)orv[i] * ACOL + n] = acc[t][i] + bcol + rbv[i];
                }
            }
        }
    }
}

// ---------------------------------------------------------------------------
// Zero only the padding rows of out (d >= count for that batch).
// ---------------------------------------------------------------------------
__global__ void k_zeropad(float* __restrict__ out, const int* __restrict__ si,
                          int b, int max_d) {
    int total = b * max_d;
    for (int r = blockIdx.x; r < total; r += gridDim.x) {
        int batch = r / max_d, d = r - batch * max_d;
        int cnt = si[batch + 1] - si[batch];
        if (d >= cnt) {
            float4 z = make_float4(0.f, 0.f, 0.f, 0.f);
            float4* p = (float4*)(out + (size_t)r * ACOL);
            for (int i = threadIdx.x; i < ACOL / 4; i += blockDim.x) p[i] = z;
        }
    }
}

extern "C" void kernel_launch(void* const* d_in, const int* in_sizes, int n_in,
                              void* d_out, int out_size, void* d_ws, size_t ws_size,
                              hipStream_t stream) {
    const float* X   = (const float*)d_in[0];
    const float* W1d = (const float*)d_in[1];
    const float* b1d = (const float*)d_in[2];
    const float* W2d = (const float*)d_in[3];
    const float* b2d = (const float*)d_in[4];
    const float* W1a = (const float*)d_in[5];
    const float* b1a = (const float*)d_in[6];
    const float* W2a = (const float*)d_in[7];
    const float* b2a = (const float*)d_in[8];
    const int*   bidx = (const int*)d_in[9];
    const int*   sidx = (const int*)d_in[10];

    const int N = in_sizes[0] / E;
    const int b = in_sizes[10] - 1;
    const int a = in_sizes[8];                 // 4096
    const int max_d = out_size / (b * a);

    // workspace layout
    char* ws = (char*)d_ws;
    size_t oWt1d = 0;
    size_t oWt1a = oWt1d + (size_t)E * E * 2;
    size_t oWt2a = oWt1a + (size_t)E * E * 2;
    size_t oDq   = oWt2a + (size_t)E * ACOL * 2;
    size_t oSeg  = oDq + (size_t)N * 2 * sizeof(float);
    size_t oXbf  = (oSeg + (size_t)b * sizeof(float) + 255) & ~(size_t)255;
    const bool useXbf = ws_size >= oXbf + (size_t)N * E * 2;

    bf16*  Wt1d = (bf16*)(ws + oWt1d);
    bf16*  Wt1a = (bf16*)(ws + oWt1a);
    bf16*  Wt2a = (bf16*)(ws + oWt2a);
    float* dq   = (float*)(ws + oDq);
    float* seg  = (float*)(ws + oSeg);
    bf16*  Xbf  = useXbf ? (bf16*)(ws + oXbf) : nullptr;

    float* out = (float*)d_out;

    k_convT<<<(E * E + 255) / 256, 256, 0, stream>>>(W1d, Wt1d, E, E);
    k_convT<<<(E * E + 255) / 256, 256, 0, stream>>>(W1a, Wt1a, E, E);
    k_convT<<<(E * ACOL + 255) / 256, 256, 0, stream>>>(W2a, Wt2a, E, ACOL);
    if (useXbf) {
        size_t total = (size_t)N * E;
        k_convX<<<(unsigned)((total + 255) / 256), 256, 0, stream>>>(X, Xbf, total);
    }

    int nblk = (N + BM - 1) / BM;
    k_device<<<nblk, 256, 0, stream>>>(X, Xbf, Wt1d, b1d, W2d, b2d, dq, N);
    k_seg<<<b, 32, 0, stream>>>(dq, sidx, seg);
    k_action<<<nblk, 256, 0, stream>>>(X, Xbf, Wt1a, b1a, Wt2a, b2a, dq, seg,
                                       bidx, sidx, out, N, max_d);
    k_zeropad<<<2048, 256, 0, stream>>>(out, sidx, b, max_d);
}
